// PerceiverAttentionCA_59356448031295
// MI455X (gfx1250) — matrix operand order repaired and checked by reading, not compiled, not run on hardware
//
#include <hip/hip_runtime.h>
#include <hip/hip_bf16.h>
#include <math.h>

// ---------------------------------------------------------------------------
// PerceiverAttentionCA for MI455X (gfx1250, wave32, WMMA bf16, async-to-LDS)
// B=4 T=32 LA=32 L=512 D=2048 H=16 C=128 KV=2048 K=1024
// ---------------------------------------------------------------------------

typedef __attribute__((ext_vector_type(16))) __bf16 v16bf;
typedef __attribute__((ext_vector_type(8)))  float  v8f;

#define BATCH   4
#define SEQ_L   512
#define SEQ_K   1024
#define DIM     2048
#define HEADS   16
#define HDIM    128
#define KVDIM   2048
#define LDT     40          // LDS row stride (elements); 80B = multiple of 16B

__device__ __forceinline__ v8f v8f_zero() {
    v8f z = {0.f,0.f,0.f,0.f,0.f,0.f,0.f,0.f};
    return z;
}

// load a 32-byte WMMA fragment as two 16-byte chunks (elements 0..7 at p0, 8..15 at p1)
__device__ __forceinline__ v16bf load2(const __bf16* p0, const __bf16* p1) {
    union { v16bf v; uint4 u[2]; } f;
    f.u[0] = *(const uint4*)p0;
    f.u[1] = *(const uint4*)p1;
    return f.v;
}

__device__ __forceinline__ v8f wmma_bf16(v16bf a, v16bf b, v8f c) {
    return __builtin_amdgcn_wmma_f32_16x16x32_bf16(
        /*neg_a=*/false, a, /*neg_b=*/false, b,
        /*c_mod=*/(short)0, c, /*reuse_a=*/false, /*reuse_b=*/false);
}

// CDNA5 async copy: global -> LDS, 16B per lane, no VGPR data traffic.
// ldsOff = LDS byte address (low 32 bits of generic shared pointer).
__device__ __forceinline__ void async_ld_b128(unsigned ldsOff, const void* g) {
    asm volatile("global_load_async_to_lds_b128 %0, %1, off"
                 :: "v"(ldsOff), "v"(g)
                 : "memory");
}
__device__ __forceinline__ void wait_asynccnt0() {
    asm volatile("s_wait_asynccnt 0x0" ::: "memory");
}
__device__ __forceinline__ unsigned lds_off(const void* p) {
    return (unsigned)(size_t)p;   // generic LDS address: low 32 bits = LDS offset
}

// ---------------------------------------------------------------------------
// LayerNorm(x) * w + b  -> bf16   (one row of 2048 per block, 256 threads)
// ---------------------------------------------------------------------------
__global__ __launch_bounds__(256)
void ln_affine_bf16(const float* __restrict__ x, const float* __restrict__ w,
                    const float* __restrict__ bvec, __bf16* __restrict__ out) {
    const int row = blockIdx.x;
    const int tid = threadIdx.x;
    const float* xr = x + (size_t)row * DIM;
    float v[8], s = 0.f, ss = 0.f;
#pragma unroll
    for (int i = 0; i < 8; ++i) {
        float t = xr[tid + i * 256];
        v[i] = t; s += t; ss += t * t;
    }
#pragma unroll
    for (int off = 16; off; off >>= 1) {
        s  += __shfl_xor(s,  off, 32);
        ss += __shfl_xor(ss, off, 32);
    }
    __shared__ float red[2][8];
    const int wave = tid >> 5, lane = tid & 31;
    if (lane == 0) { red[0][wave] = s; red[1][wave] = ss; }
    __syncthreads();
    s = 0.f; ss = 0.f;
#pragma unroll
    for (int i = 0; i < 8; ++i) { s += red[0][i]; ss += red[1][i]; }
    const float mean = s * (1.f / DIM);
    const float var  = ss * (1.f / DIM) - mean * mean;
    const float rstd = rsqrtf(var + 1e-5f);
    __bf16* orow = out + (size_t)row * DIM;
#pragma unroll
    for (int i = 0; i < 8; ++i) {
        int c = tid + i * 256;
        orow[c] = (__bf16)((v[i] - mean) * rstd * w[c] + bvec[c]);
    }
}

// ---------------------------------------------------------------------------
// LayerNorm(latents) * (1 + scale[b]) + shift[b] -> bf16 (scale/shift = t_emb+ssg)
// ---------------------------------------------------------------------------
__global__ __launch_bounds__(256)
void ln_mod_bf16(const float* __restrict__ lat, const float* __restrict__ t_emb,
                 const float* __restrict__ ssg, __bf16* __restrict__ out) {
    const int row = blockIdx.x;          // b*L + l
    const int b   = row >> 9;            // L = 512
    const int tid = threadIdx.x;
    const float* xr = lat + (size_t)row * DIM;
    float v[8], s = 0.f, ss = 0.f;
#pragma unroll
    for (int i = 0; i < 8; ++i) {
        float t = xr[tid + i * 256];
        v[i] = t; s += t; ss += t * t;
    }
#pragma unroll
    for (int off = 16; off; off >>= 1) {
        s  += __shfl_xor(s,  off, 32);
        ss += __shfl_xor(ss, off, 32);
    }
    __shared__ float red[2][8];
    const int wave = tid >> 5, lane = tid & 31;
    if (lane == 0) { red[0][wave] = s; red[1][wave] = ss; }
    __syncthreads();
    s = 0.f; ss = 0.f;
#pragma unroll
    for (int i = 0; i < 8; ++i) { s += red[0][i]; ss += red[1][i]; }
    const float mean = s * (1.f / DIM);
    const float var  = ss * (1.f / DIM) - mean * mean;
    const float rstd = rsqrtf(var + 1e-5f);
    __bf16* orow = out + (size_t)row * DIM;
#pragma unroll
    for (int i = 0; i < 8; ++i) {
        int c = tid + i * 256;
        float scale = t_emb[((size_t)b * 3 + 1) * DIM + c] + ssg[DIM + c];
        float shift = t_emb[((size_t)b * 3 + 0) * DIM + c] + ssg[c];
        orow[c] = (__bf16)(((v[i] - mean) * rstd) * (1.f + scale) + shift);
    }
}

// ---------------------------------------------------------------------------
// fp32 -> bf16 convert (grid-stride)
// ---------------------------------------------------------------------------
__global__ __launch_bounds__(256)
void cvt_bf16(const float* __restrict__ src, __bf16* __restrict__ dst, int n) {
    for (int i = blockIdx.x * blockDim.x + threadIdx.x; i < n;
         i += gridDim.x * blockDim.x)
        dst[i] = (__bf16)src[i];
}

// ---------------------------------------------------------------------------
// WMMA bf16 GEMM:  out[M,N] = A[M,Kd] @ Bm[Kd,N] + bias
//   MODE 0: store bf16
//   MODE 1: store fp32, multiplied by gate[b,col] = t_emb[b,2,:]+ssg[0,2,:]
// Block: 256 threads (8 waves), tile 128x128, BK=32. Wave = 32 rows x 64 cols.
// A tile staged via GLOBAL_LOAD_ASYNC_TO_LDS_B128 (ASYNCcnt, zero VGPR traffic);
// B tile staged through VGPRs because it is transposed on the LDS write.
// ---------------------------------------------------------------------------
template <int MODE>
__global__ __launch_bounds__(256)
void wmma_gemm_bf16(const __bf16* __restrict__ A, const __bf16* __restrict__ Bm,
                    const float* __restrict__ bias,
                    __bf16* __restrict__ outBf, float* __restrict__ outF,
                    const float* __restrict__ t_emb, const float* __restrict__ ssg,
                    int M, int N, int Kd, int rows_per_batch) {
    __shared__ __bf16 Asub[128 * LDT];   // [row][k]
    __shared__ __bf16 Btsub[128 * LDT];  // transposed: [col][k]

    const int tid   = threadIdx.x;
    const int lane  = tid & 31;
    const int wave  = tid >> 5;          // 0..7
    const int waveM = wave >> 1;         // 0..3  (32 rows each)
    const int waveN = wave & 1;          // 0..1  (64 cols each)
    const int hi    = lane >> 4;
    const int l15   = lane & 15;

    const int rowBase = blockIdx.y * 128;
    const int colBase = blockIdx.x * 128;

    // per-thread staging coordinates
    const int ar = tid >> 1;                  // A: row 0..127
    const int ac = (tid & 1) * 16;            // A: col 0 or 16
    const int br = tid >> 3;                  // B: row 0..31
    const int bc = (tid & 7) * 16;            // B: col 0..112
    const __bf16* Ag = A + (size_t)(rowBase + ar) * Kd + ac;
    const __bf16* Bg = Bm + (size_t)br * N + colBase + bc;
    const unsigned aLds = lds_off(Asub + ar * LDT + ac);

    v8f acc[2][4];
#pragma unroll
    for (int mt = 0; mt < 2; ++mt)
#pragma unroll
        for (int nt = 0; nt < 4; ++nt) acc[mt][nt] = v8f_zero();

    for (int kb = 0; kb < Kd; kb += 32) {
        __syncthreads();
        // ---- A tile (128x32): async DMA global->LDS, 2 x b128 per thread ----
        async_ld_b128(aLds,      Ag + kb);
        async_ld_b128(aLds + 16, Ag + kb + 8);
        // ---- prefetch next iteration's tiles into GL2 ----
        if (kb + 32 < Kd) {
            __builtin_prefetch(Ag + kb + 32, 0, 1);
            __builtin_prefetch(Bg + (size_t)(kb + 32) * N, 0, 1);
        }
        // ---- B tile (32x128) transposed -> Btsub[col][k] ----
        {
            const uint4* g = (const uint4*)(Bg + (size_t)kb * N);
            union { uint4 u[2]; __bf16 e[16]; } t;
            t.u[0] = g[0]; t.u[1] = g[1];
#pragma unroll
            for (int e = 0; e < 16; ++e) Btsub[(bc + e) * LDT + br] = t.e[e];
        }
        wait_asynccnt0();
        __syncthreads();

        v16bf af[2], bfr[4];
#pragma unroll
        for (int mt = 0; mt < 2; ++mt) {
            // A frag 16x32: lane lo-half K 0..7/16..23, hi-half K 8..15/24..31
            const __bf16* p = Asub + (waveM * 32 + mt * 16 + l15) * LDT + hi * 8;
            af[mt] = load2(p, p + 16);
        }
#pragma unroll
        for (int nt = 0; nt < 4; ++nt) {
            // B frag 32x16: lane lo-half K 0..7/16..23, hi-half K 8..15/24..31
            const __bf16* p = Btsub + (waveN * 64 + nt * 16 + l15) * LDT + hi * 8;
            bfr[nt] = load2(p, p + 16);
        }
#pragma unroll
        for (int mt = 0; mt < 2; ++mt)
#pragma unroll
            for (int nt = 0; nt < 4; ++nt)
                acc[mt][nt] = wmma_bf16(af[mt], bfr[nt], acc[mt][nt]);
    }

    // epilogue: D layout -> row = base + j + 8*hi, col = base + (lane&15)
#pragma unroll
    for (int mt = 0; mt < 2; ++mt) {
#pragma unroll
        for (int nt = 0; nt < 4; ++nt) {
            const int col = colBase + waveN * 64 + nt * 16 + l15;
#pragma unroll
            for (int j = 0; j < 8; ++j) {
                const int row = rowBase + waveM * 32 + mt * 16 + j + hi * 8;
                float v = acc[mt][nt][j] + bias[col];
                if (MODE == 0) {
                    outBf[(size_t)row * N + col] = (__bf16)v;
                } else {
                    const int b = row / rows_per_batch;
                    const float gate = t_emb[((size_t)b * 3 + 2) * N + col] +
                                       ssg[(size_t)2 * N + col];
                    outF[(size_t)row * N + col] = v * gate;
                }
            }
        }
    }
}

// ---------------------------------------------------------------------------
// Flash attention.  grid = (B*H, L/64), block = 128 (4 waves).
// Each wave: one 16-row Q tile, streams K in chunks of 32 columns.
//   S tile (16x32) via 8 WMMAs over C=128; online softmax in D layout
//   (row reductions = width-16 shuffles); P -> per-wave LDS -> A layout;
//   PV via 8 WMMAs against V staged transposed in LDS.
// kv buffer rows: [k(0..2047) | v(2048..4095)] per (b*K + kidx).
// ---------------------------------------------------------------------------
__global__ __launch_bounds__(128)
void attn_fa(const __bf16* __restrict__ qb, const __bf16* __restrict__ kvb,
             const int* __restrict__ k_lens, const int* __restrict__ q_lens,
             __bf16* __restrict__ attnOut) {
    __shared__ __bf16 Vt[HDIM * LDT];          // transposed V tile: [c][k]
    __shared__ __bf16 Plds[4 * 16 * LDT];      // per-wave P scratch

    const int tid  = threadIdx.x;
    const int lane = tid & 31;
    const int wave = tid >> 5;                 // 0..3
    const int hi   = lane >> 4;
    const int l15  = lane & 15;

    const int bh = blockIdx.x;
    const int b  = bh >> 4;                    // H = 16
    const int h  = bh & 15;
    const int qRow0 = blockIdx.y * 64 + wave * 16;

    const int klen = k_lens[b];
    const int kmax = (klen + 31) & ~31;

    // Q A-fragments for the whole C=128, loaded once from global (row-major).
    v16bf qf[4];
    {
        const __bf16* qrow = qb + ((size_t)(b * SEQ_L + qRow0 + l15)) * DIM +
                             h * HDIM + hi * 8;
#pragma unroll
        for (int cc = 0; cc < 4; ++cc) {
            const __bf16* p = qrow + cc * 32;
            qf[cc] = load2(p, p + 16);
        }
    }

    v8f acc[8];
#pragma unroll
    for (int t = 0; t < 8; ++t) acc[t] = v8f_zero();
    float m8[8], l8[8];
#pragma unroll
    for (int j = 0; j < 8; ++j) { m8[j] = -__builtin_inff(); l8[j] = 0.f; }

    const size_t kvbase = (size_t)b * SEQ_K * 4096;
    const float sc = 0.08838834764831845f;     // 1/sqrt(128)

    for (int kb = 0; kb < kmax; kb += 32) {
        __syncthreads();
        {   // stage V tile (32 rows x 128 c) transposed into Vt[c][k]
            int r  = tid >> 2;                 // 0..31
            int c0 = (tid & 3) * 32;           // 0,32,64,96
            const uint4* g = (const uint4*)(kvb + kvbase +
                             (size_t)(kb + r) * 4096 + DIM + h * HDIM + c0);
            union { uint4 u[4]; __bf16 e[32]; } t;
            t.u[0] = g[0]; t.u[1] = g[1]; t.u[2] = g[2]; t.u[3] = g[3];
#pragma unroll
            for (int e = 0; e < 32; ++e) Vt[(c0 + e) * LDT + r] = t.e[e];
        }
        __syncthreads();

        // ---- S = Q * K^T  (two 16x16 n-tiles covering k cols kb..kb+31) ----
        v8f s0 = v8f_zero(), s1 = v8f_zero();
#pragma unroll
        for (int cc = 0; cc < 4; ++cc) {
            // K^T B-frag: row-major K means lane reads contiguous c from k-row
            const __bf16* kr0 = kvb + kvbase + (size_t)(kb + l15) * 4096 +
                                h * HDIM + cc * 32 + hi * 8;
            const __bf16* kr1 = kr0 + (size_t)16 * 4096;
            v16bf b0 = load2(kr0, kr0 + 16);
            v16bf b1 = load2(kr1, kr1 + 16);
            s0 = wmma_bf16(qf[cc], b0, s0);
            s1 = wmma_bf16(qf[cc], b1, s1);
        }

        // ---- online softmax in D layout ----
        const int col0 = kb + l15, col1 = col0 + 16;
        float corr8[8];
        __bf16* pl = Plds + wave * 16 * LDT;
#pragma unroll
        for (int j = 0; j < 8; ++j) {
            float a0 = (col0 < klen) ? s0[j] * sc : -__builtin_inff();
            float a1 = (col1 < klen) ? s1[j] * sc : -__builtin_inff();
            float mloc = fmaxf(a0, a1);
#pragma unroll
            for (int off = 8; off; off >>= 1)
                mloc = fmaxf(mloc, __shfl_xor(mloc, off, 16));
            float mnew = fmaxf(m8[j], mloc);
            float corr = __expf(m8[j] - mnew);
            float p0 = __expf(a0 - mnew);
            float p1 = __expf(a1 - mnew);
            float rs = p0 + p1;
#pragma unroll
            for (int off = 8; off; off >>= 1) rs += __shfl_xor(rs, off, 16);
            l8[j] = l8[j] * corr + rs;
            m8[j] = mnew;
            corr8[j] = corr;
            const int prow = j + hi * 8;
            pl[prow * LDT + l15]      = (__bf16)p0;
            pl[prow * LDT + 16 + l15] = (__bf16)p1;
        }
#pragma unroll
        for (int t = 0; t < 8; ++t)
#pragma unroll
            for (int j = 0; j < 8; ++j) acc[t][j] *= corr8[j];

        // P as A-fragment (same-wave LDS RAW: DS ops are in-order per wave)
        const __bf16* pp = pl + l15 * LDT + hi * 8;
        v16bf pf = load2(pp, pp + 16);

        // ---- O += P * V ----
#pragma unroll
        for (int ct = 0; ct < 8; ++ct) {
            const __bf16* vp = Vt + (ct * 16 + l15) * LDT + hi * 8;
            v16bf vf = load2(vp, vp + 16);
            acc[ct] = wmma_bf16(pf, vf, acc[ct]);
        }
    }

    // ---- normalize, apply q mask, store bf16 ----
    const int qlen = q_lens[b];
    float inv8[8];
#pragma unroll
    for (int j = 0; j < 8; ++j) inv8[j] = 1.f / l8[j];
#pragma unroll
    for (int ct = 0; ct < 8; ++ct) {
#pragma unroll
        for (int j = 0; j < 8; ++j) {
            const int lrow = qRow0 + j + hi * 8;
            float o = acc[ct][j] * inv8[j];
            if (lrow >= qlen) o = 0.f;
            attnOut[((size_t)(b * SEQ_L + lrow)) * DIM +
                    h * HDIM + ct * 16 + l15] = (__bf16)o;
        }
    }
}

// ---------------------------------------------------------------------------
// Host orchestration
// ---------------------------------------------------------------------------
extern "C" void kernel_launch(void* const* d_in, const int* in_sizes, int n_in,
                              void* d_out, int out_size, void* d_ws, size_t ws_size,
                              hipStream_t stream) {
    const float* x        = (const float*)d_in[0];
    const float* latents  = (const float*)d_in[1];
    const float* t_emb    = (const float*)d_in[2];
    const int*   q_lens   = (const int*)d_in[3];
    const int*   k_lens   = (const int*)d_in[4];
    const float* ln_kv_w  = (const float*)d_in[5];
    const float* ln_kv_b  = (const float*)d_in[6];
    const float* ssg      = (const float*)d_in[7];
    const float* Wq       = (const float*)d_in[8];
    const float* bq       = (const float*)d_in[9];
    const float* Wkv      = (const float*)d_in[10];
    const float* bkv      = (const float*)d_in[11];
    const float* Wo       = (const float*)d_in[12];
    const float* bo       = (const float*)d_in[13];
    float* out = (float*)d_out;

    char* ws = (char*)d_ws;
    __bf16* latb  = (__bf16*)ws; ws += (size_t)BATCH * SEQ_L * DIM * 2;   // 8 MB
    __bf16* xkb   = (__bf16*)ws; ws += (size_t)BATCH * SEQ_K * KVDIM * 2; // 16 MB
    __bf16* Wqb   = (__bf16*)ws; ws += (size_t)DIM * DIM * 2;             // 8 MB
    __bf16* Wkvb  = (__bf16*)ws; ws += (size_t)KVDIM * 2 * DIM * 2;       // 16 MB
    __bf16* Wob   = (__bf16*)ws; ws += (size_t)DIM * DIM * 2;             // 8 MB
    __bf16* qbuf  = (__bf16*)ws; ws += (size_t)BATCH * SEQ_L * DIM * 2;   // 8 MB
    __bf16* kvbuf = (__bf16*)ws; ws += (size_t)BATCH * SEQ_K * 4096 * 2;  // 32 MB
    __bf16* attnb = (__bf16*)ws;                                          // 8 MB

    // 1) LN(x)*w+b -> bf16
    ln_affine_bf16<<<BATCH * SEQ_K, 256, 0, stream>>>(x, ln_kv_w, ln_kv_b, xkb);
    // 2) LN(latents) modulated -> bf16
    ln_mod_bf16<<<BATCH * SEQ_L, 256, 0, stream>>>(latents, t_emb, ssg, latb);
    // 3) weights -> bf16
    cvt_bf16<<<2048, 256, 0, stream>>>(Wq,  Wqb,  DIM * DIM);
    cvt_bf16<<<2048, 256, 0, stream>>>(Wkv, Wkvb, KVDIM * 2 * DIM);
    cvt_bf16<<<2048, 256, 0, stream>>>(Wo,  Wob,  DIM * DIM);
    // 4) q = lat @ Wq + bq            (M=2048, N=2048, K=2048)
    wmma_gemm_bf16<0><<<dim3(16, 16), 256, 0, stream>>>(
        latb, Wqb, bq, qbuf, nullptr, nullptr, nullptr, 2048, 2048, 2048, SEQ_L);
    // 5) kv = xk @ Wkv + bkv          (M=4096, N=4096, K=2048)
    wmma_gemm_bf16<0><<<dim3(32, 32), 256, 0, stream>>>(
        xkb, Wkvb, bkv, kvbuf, nullptr, nullptr, nullptr, 4096, 4096, 2048, SEQ_K);
    // 6) flash attention
    attn_fa<<<dim3(BATCH * HEADS, SEQ_L / 64), 128, 0, stream>>>(
        qbuf, kvbuf, k_lens, q_lens, attnb);
    // 7) out = (attn @ Wo + bo) * gate  -> fp32
    wmma_gemm_bf16<1><<<dim3(16, 16), 256, 0, stream>>>(
        attnb, Wob, bo, nullptr, out, t_emb, ssg, 2048, 2048, 2048, SEQ_L);
}
